// MultiHeadLstm_49563922596235
// MI455X (gfx1250) — matrix operand order repaired and checked by reading, not compiled, not run on hardware
//
#include <hip/hip_runtime.h>
#include <hip/hip_bf16.h>
#include <math.h>

typedef __attribute__((ext_vector_type(16))) _Float16 v16h;
typedef __attribute__((ext_vector_type(8)))  float    v8f;

#define B_SZ   64
#define S_LEN  512
#define I_DIM  768
#define NHEAD  12
#define H_DIM  768
#define MROWS  (B_SZ * S_LEN)                 // 32768
#define OUT_ELEMS ((long)MROWS * I_DIM)       // 25165824

// ---------------------------------------------------------------------------
// Branchless fast transcendentals (v_exp_f32 + v_rcp_f32, no libm branches).
//   sigmoid(x) = rcp(1 + exp2(-log2(e)*x))
//   tanh(x)    = 2*rcp(1 + exp2(-2*log2(e)*x)) - 1
// ---------------------------------------------------------------------------
#define LOG2E  1.442695041f
static __device__ __forceinline__ float fast_rcp(float x) {
  return __builtin_amdgcn_rcpf(x);
}
static __device__ __forceinline__ float fast_exp2(float x) {
  return __builtin_amdgcn_exp2f(x);
}
static __device__ __forceinline__ float fast_tanh(float x) {
  return fmaf(2.0f, fast_rcp(1.0f + fast_exp2(-2.0f * LOG2E * x)), -1.0f);
}

// ---------------------------------------------------------------------------
// WMMA operand loaders for v_wmma_f32_16x16x32_f16 (wave32).
// A (16x32 f16): lane L<16 holds row M=L, K = 0..7 (v0-3) & 16..23 (v4-7);
//                lane L>=16 holds row M=L-16, K = 8..15 & 24..31.
// B (32x16 f16): lane L<16 holds col N=L, K = 0..7 (v0-3) & 16..23 (v4-7);
//                lane L>=16 holds col N=L-16, K = 8..15 & 24..31.
// C/D (16x16 f32, 8 VGPRs): vgpr r -> row r (lanes 0-15) / row r+8 (lanes 16-31),
//                col = lane & 15.
// ---------------------------------------------------------------------------
template<typename T>
static __device__ __forceinline__ v16h load_a16(const T* src, long ldr) {
  const int lane = threadIdx.x & 31;
  const int m    = lane & 15;
  const int kb   = (lane >> 4) << 3;
  const T* p = src + (long)m * ldr + kb;
  v16h a;
#pragma unroll
  for (int j = 0; j < 8; ++j) {
    a[j]     = (_Float16)p[j];         // K = kb + j
    a[j + 8] = (_Float16)p[j + 16];    // K = 16 + kb + j
  }
  return a;
}

// B[k][n] = W[n][k] for row-major W (i.e. this computes A @ W^T).
static __device__ __forceinline__ v16h load_bT(const float* W, long ldw) {
  const int lane = threadIdx.x & 31;
  const int n    = lane & 15;
  const int kh   = (lane >> 4) << 3;
  const float* p = W + (long)n * ldw + kh;
  v16h b;
#pragma unroll
  for (int j = 0; j < 8; ++j) {
    b[j]     = (_Float16)p[j];         // K = kh + j
    b[j + 8] = (_Float16)p[j + 16];    // K = 16 + kh + j
  }
  return b;
}

static __device__ __forceinline__ v8f wmma_f16(v16h a, v16h b, v8f c) {
  return __builtin_amdgcn_wmma_f32_16x16x32_f16(false, a, false, b, (short)0, c,
                                                false, false);
}

// ---------------------------------------------------------------------------
// Dense GEMM: OUT[M,N] = A[M,K] @ W[N,K]^T + bias.  Block = 8 waves covering a
// 128(M) x 64(N) tile; each wave owns one 16-row M-tile and four N-tiles.
// ---------------------------------------------------------------------------
template<typename AT>
__global__ __launch_bounds__(256)
void gemm_wt_kernel(const AT* __restrict__ A, const float* __restrict__ W,
                    const float* __restrict__ bias,
                    float* __restrict__ outF, _Float16* __restrict__ outH,
                    int M, int N, int K) {
  const int wave = threadIdx.x >> 5;
  const int lane = threadIdx.x & 31;
  const int nloc = lane & 15;
  const int mh   = (lane >> 4) << 3;
  const int m0 = blockIdx.y * 128 + wave * 16;
  const int n0 = blockIdx.x * 64;

  v8f acc[4];
#pragma unroll
  for (int t = 0; t < 4; ++t) {
    const float bv = bias[n0 + t * 16 + nloc];
#pragma unroll
    for (int r = 0; r < 8; ++r) acc[t][r] = bv;
  }

  for (int k0 = 0; k0 < K; k0 += 32) {
    v16h a = load_a16(A + (long)m0 * K + k0, K);
#pragma unroll
    for (int t = 0; t < 4; ++t) {
      v16h b = load_bT(W + (long)(n0 + t * 16) * K + k0, K);
      acc[t] = wmma_f16(a, b, acc[t]);
    }
  }

#pragma unroll
  for (int t = 0; t < 4; ++t)
#pragma unroll
    for (int r = 0; r < 8; ++r) {
      const long row = m0 + mh + r;
      const int  col = n0 + t * 16 + nloc;
      if (outF) outF[row * N + col] = acc[t][r];
      if (outH) outH[row * N + col] = (_Float16)acc[t][r];
    }
}

// ---------------------------------------------------------------------------
// Bidirectional multi-head LSTM scan. One block per (dir, head, batch-quarter):
// 96 independent sequential chains. Per step: G[16,256] = [x_head|h] @ [Wih|Whh]^T
// with the 256x128 gate-weight block resident in VGPRs (8 v16h tiles / wave).
// Gate order i,f,g,o. h/c state in LDS. x A-tiles are prefetched one step
// ahead; all per-step addressing is running-pointer based; accumulators start
// from an inline-zero C operand and the bias is folded into the exp2 argument.
// ---------------------------------------------------------------------------
__global__ __launch_bounds__(256)
void lstm_scan_kernel(const _Float16* __restrict__ xpre,
                      const float* __restrict__ wih_f, const float* __restrict__ whh_f,
                      const float* __restrict__ bih_f, const float* __restrict__ bhh_f,
                      const float* __restrict__ wih_b, const float* __restrict__ whh_b,
                      const float* __restrict__ bih_b, const float* __restrict__ bhh_b,
                      _Float16* __restrict__ hbuf) {
  const int bq    = blockIdx.x & 3;        // batch quarter: rows 16*bq..16*bq+15
  const int chain = blockIdx.x >> 2;       // 0..23
  const int dir   = chain / NHEAD;
  const int head  = chain % NHEAD;
  const float* wih = dir ? wih_b : wih_f;
  const float* whh = dir ? whh_b : whh_f;
  const float* bih = dir ? bih_b : bih_f;
  const float* bhh = dir ? bhh_b : bhh_f;

  __shared__ _Float16 hsh[16][64];         // h_{t-1} as f16 A-operand
  __shared__ float    gact[16][256];       // activated gates
  __shared__ float    csh[16][64];         // cell state

  const int tid  = threadIdx.x;
  const int wave = tid >> 5;
  const int lane = tid & 31;
  const int nloc = lane & 15;
  const int mh   = (lane >> 4) << 3;

  for (int i = tid; i < 16 * 64; i += 256) {
    csh[i >> 6][i & 63] = 0.0f;
    hsh[i >> 6][i & 63] = (_Float16)0.0f;
  }

  // Resident weights: wave owns gate-cols 32*wave..32*wave+31 (N-tiles 2w,2w+1).
  // K chunks 0,1 = x part (wih), chunks 2,3 = h part (whh).
  v16h  Bt[2][4];
  float biasv[2];
#pragma unroll
  for (int s = 0; s < 2; ++s) {
    const int n0   = (wave * 2 + s) * 16;
    const int grow = (n0 >> 6) * H_DIM + head * 64 + (n0 & 63);  // global weight row
#pragma unroll
    for (int c = 0; c < 4; ++c) {
      const float* W = (c < 2) ? wih : whh;
      const int kc = (c & 1) * 32;
      Bt[s][c] = load_bT(W + (long)grow * I_DIM + head * 64 + kc, I_DIM);
    }
    const int n  = n0 + nloc;
    const int bi = (n >> 6) * H_DIM + head * 64 + (n & 63);
    biasv[s] = bih[bi] + bhh[bi];
  }

  // Per-gate activation selectors, wave-uniform, hoisted out of the time loop:
  //   gates i,f,o: sigmoid(x) = rcp(1 + exp2(-log2e*x))        (mul=1, add=0)
  //   gate  g    : tanh(x)    = 2*rcp(1 + exp2(-2*log2e*x)) - 1 (mul=2, add=-1)
  // Bias folded into the exp2 argument: exp2(escl*g + escl*bias).
  const int   gateg = wave >> 1;           // 0=i 1=f 2=g 3=o
  const float escl  = (gateg == 2) ? (-2.0f * LOG2E) : (-LOG2E);
  const float vmul  = (gateg == 2) ? 2.0f : 1.0f;
  const float vadd  = (gateg == 2) ? -1.0f : 0.0f;
  const float ebias[2] = { escl * biasv[0], escl * biasv[1] };

  // Running pointers (no per-step index re-derivation).
  const long xrow  = (long)S_LEN * I_DIM;          // batch-row stride in xpre
  const long hrow  = (long)S_LEN * (2 * H_DIM);    // batch-row stride in hbuf
  const long xstep = dir ? -(long)I_DIM : (long)I_DIM;
  const long hstep = dir ? -(long)(2 * H_DIM) : (long)(2 * H_DIM);
  const int  tr0   = dir ? (S_LEN - 1) : 0;
  const _Float16* xptr = xpre + (long)(bq * 16) * xrow + (long)tr0 * I_DIM + head * 64;
  _Float16*       hptr = hbuf + (long)(bq * 16) * hrow + (long)tr0 * (2 * H_DIM)
                              + dir * H_DIM + head * 64;

  __syncthreads();

  // Prefetch x A-tiles for step 0.
  v16h ax0 = load_a16(xptr, xrow);
  v16h ax1 = load_a16(xptr + 32, xrow);

  for (int t = 0; t < S_LEN; ++t) {
    // x contribution: first WMMA pair starts from inline-zero C (no mov chain).
    v8f zero;
#pragma unroll
    for (int r = 0; r < 8; ++r) zero[r] = 0.0f;
    v8f acc[2];
#pragma unroll
    for (int s = 0; s < 2; ++s) acc[s] = wmma_f16(ax0, Bt[s][0], zero);
#pragma unroll
    for (int s = 0; s < 2; ++s) acc[s] = wmma_f16(ax1, Bt[s][1], acc[s]);

    // h_{t-1} contribution from LDS (both chunks loaded up front -> one clause)
    {
      v16h ah0 = load_a16(&hsh[0][0], 64);
      v16h ah1 = load_a16(&hsh[0][0] + 32, 64);
#pragma unroll
      for (int s = 0; s < 2; ++s) acc[s] = wmma_f16(ah0, Bt[s][2], acc[s]);
#pragma unroll
      for (int s = 0; s < 2; ++s) acc[s] = wmma_f16(ah1, Bt[s][3], acc[s]);
    }

    // Prefetch next step's x tiles (clamped on the final step so the backward
    // direction never reads below the buffer base); latency hides behind the
    // barriers + cell update.
    xptr += (t + 1 < S_LEN) ? xstep : 0;
    ax0 = load_a16(xptr, xrow);
    ax1 = load_a16(xptr + 32, xrow);

    // Activations (branchless, bias fused into the FMA feeding v_exp_f32).
#pragma unroll
    for (int s = 0; s < 2; ++s) {
      const int n = wave * 32 + s * 16 + nloc;
#pragma unroll
      for (int r = 0; r < 8; ++r) {
        const float e  = fast_exp2(fmaf(escl, acc[s][r], ebias[s]));
        const float rv = fast_rcp(1.0f + e);
        gact[mh + r][n] = fmaf(vmul, rv, vadd);
      }
    }
    __syncthreads();

    // Cell update + emit h (f16) in [b,s, dir*H + head*64 + j] concat layout.
    for (int i = tid; i < 16 * 64; i += 256) {
      const int b = i >> 6, j = i & 63;
      const float ig = gact[b][j];
      const float fg = gact[b][64 + j];
      const float gg = gact[b][128 + j];
      const float og = gact[b][192 + j];
      const float cn = fmaf(fg, csh[b][j], ig * gg);
      csh[b][j] = cn;
      const float hn = og * fast_tanh(cn);
      hsh[b][j] = (_Float16)hn;
      hptr[(long)b * hrow + j] = (_Float16)hn;
    }
    hptr += hstep;   // final advance never dereferenced
    __syncthreads();
  }
}

// ---------------------------------------------------------------------------
// Block-diagonal head projection: per head, gather K=128 (fwd 64 | bwd 64)
// cols of hbuf and of head_w; N=64 per head. Writes f32 (output #2) + f16.
// ---------------------------------------------------------------------------
__global__ __launch_bounds__(256)
void head_proj_kernel(const _Float16* __restrict__ hbuf,
                      const float* __restrict__ head_w,
                      const float* __restrict__ head_b,
                      float* __restrict__ outF, _Float16* __restrict__ outH) {
  const int head = blockIdx.x;
  const int wave = threadIdx.x >> 5;
  const int lane = threadIdx.x & 31;
  const int nloc = lane & 15;
  const int mh   = (lane >> 4) << 3;
  const int m0 = blockIdx.y * 128 + wave * 16;

  v8f acc[4];
#pragma unroll
  for (int t = 0; t < 4; ++t) {
    const float bv = head_b[head * 64 + t * 16 + nloc];
#pragma unroll
    for (int r = 0; r < 8; ++r) acc[t][r] = bv;
  }

#pragma unroll
  for (int c = 0; c < 4; ++c) {
    const int colbase = (c < 2) ? (head * 64 + 32 * c)
                                : (H_DIM + head * 64 + 32 * (c - 2));
    v16h a = load_a16(hbuf + (long)m0 * (2 * H_DIM) + colbase, 2 * H_DIM);
#pragma unroll
    for (int t = 0; t < 4; ++t) {
      v16h b = load_bT(head_w + (long)(head * 64 + t * 16) * (2 * H_DIM) + colbase,
                       2 * H_DIM);
      acc[t] = wmma_f16(a, b, acc[t]);
    }
  }

#pragma unroll
  for (int t = 0; t < 4; ++t)
#pragma unroll
    for (int r = 0; r < 8; ++r) {
      const long row = m0 + mh + r;
      const int  col = head * 64 + t * 16 + nloc;
      outF[row * H_DIM + col] = acc[t][r];
      outH[row * H_DIM + col] = (_Float16)acc[t][r];
    }
}

// ---------------------------------------------------------------------------
// Launch. Workspace layout (f16): [xpre / lstm_head16 : 25165824][hbuf : 50331648]
// = 144 MB total. d_out = out[B,S,I] f32 then lstm_out[B,S,H] f32.
// ---------------------------------------------------------------------------
extern "C" void kernel_launch(void* const* d_in, const int* in_sizes, int n_in,
                              void* d_out, int out_size, void* d_ws, size_t ws_size,
                              hipStream_t stream) {
  const float* x      = (const float*)d_in[0];
  const float* pre_w  = (const float*)d_in[1];
  const float* pre_b  = (const float*)d_in[2];
  const float* wih_f  = (const float*)d_in[3];
  const float* whh_f  = (const float*)d_in[4];
  const float* bih_f  = (const float*)d_in[5];
  const float* bhh_f  = (const float*)d_in[6];
  const float* wih_b  = (const float*)d_in[7];
  const float* whh_b  = (const float*)d_in[8];
  const float* bih_b  = (const float*)d_in[9];
  const float* bhh_b  = (const float*)d_in[10];
  const float* head_w = (const float*)d_in[11];
  const float* head_b = (const float*)d_in[12];
  const float* post_w = (const float*)d_in[13];
  const float* post_b = (const float*)d_in[14];

  float* out      = (float*)d_out;
  float* lstm_out = out + OUT_ELEMS;

  _Float16* xpre    = (_Float16*)d_ws;       // [32768,768]; reused for head f16 out
  _Float16* hbuf    = xpre + OUT_ELEMS;      // [32768,1536]
  _Float16* lstmh16 = xpre;

  // 1) pre_proj: x @ pre_w^T + pre_b  -> f16
  gemm_wt_kernel<float><<<dim3(I_DIM / 64, MROWS / 128), 256, 0, stream>>>(
      x, pre_w, pre_b, nullptr, xpre, MROWS, I_DIM, I_DIM);

  // 2) 96 independent LSTM chains (12 heads x 2 dirs x 4 batch quarters)
  lstm_scan_kernel<<<dim3(96), 256, 0, stream>>>(
      xpre, wih_f, whh_f, bih_f, bhh_f, wih_b, whh_b, bih_b, bhh_b, hbuf);

  // 3) head projection (block-diagonal over heads) -> lstm_out f32 + f16 copy
  head_proj_kernel<<<dim3(NHEAD, MROWS / 128), 256, 0, stream>>>(
      hbuf, head_w, head_b, lstm_out, lstmh16);

  // 4) post_proj: lstm_head @ post_w^T + post_b -> out f32
  gemm_wt_kernel<_Float16><<<dim3(I_DIM / 64, MROWS / 128), 256, 0, stream>>>(
      lstmh16, post_w, post_b, out, nullptr, MROWS, I_DIM, I_DIM);

  (void)in_sizes; (void)n_in; (void)out_size; (void)ws_size;
}
